// PacketGRU_31190052504109
// MI455X (gfx1250) — compile-verified, matrix operand order checked by reading, not run on hardware
//
#include <hip/hip_runtime.h>

#define NF  16384
#define HH  64
#define TH3 192   // 3*H

typedef __attribute__((ext_vector_type(2)))  float    v2f;
typedef __attribute__((ext_vector_type(8)))  float    v8f;
typedef __attribute__((ext_vector_type(16))) _Float16 v16h;
typedef __attribute__((ext_vector_type(4)))  int      v4i;

#if __has_builtin(__builtin_amdgcn_global_load_async_to_lds_b128) && \
    __has_builtin(__builtin_amdgcn_s_wait_asynccnt)
#define USE_ASYNC_LDS 1
#endif

// ---------------------------------------------------------------------------
// K1: HBM-bound streaming kernel. One 192-thread block (6 waves) per feature.
// Half-wave-per-row: lanes 0-15 -> row 2p, lanes 16-31 -> row 2p+1, 4 h-values
// per lane. Adjacent rows are contiguous, so each wave b128 load is a fully
// coalesced 512B block (4 cachelines/instr instead of 32 with thread-per-row).
// Row dot finishes with 4 shfl_xor steps confined to each 16-lane half.
// Each lane's Ht chunk is loop-invariant: one ds_load_b128 per thread total.
// ---------------------------------------------------------------------------
__global__ __launch_bounds__(TH3) void gru_stream(
    const float* __restrict__ X, const unsigned char* __restrict__ mask,
    const float* __restrict__ Ht, const float* __restrict__ xT_w,
    const float* __restrict__ xT_b, const float* __restrict__ U_w,
    float* __restrict__ Hout)
{
  const int f = blockIdx.x;
  const int t = threadIdx.x;
  __shared__ float sHt[HH];
  __shared__ float sU[TH3];
  __shared__ float sX[TH3];

#ifdef USE_ASYNC_LDS
  if (t < 16) {
    typedef __attribute__((address_space(1))) v4i* g4p;   // global int4*
    typedef __attribute__((address_space(3))) v4i* l4p;   // LDS int4*
    __builtin_amdgcn_global_load_async_to_lds_b128(
        (g4p)(Ht + (size_t)f * HH + t * 4), (l4p)(&sHt[t * 4]), 0, 0);
  }
  __builtin_amdgcn_s_wait_asynccnt(0);   // no-op for non-issuing waves
#else
  if (t < HH) sHt[t] = Ht[(size_t)f * HH + t];
#endif
  // Warm the x-path cachelines while the big U_w stream runs.
  __builtin_prefetch(xT_w + (size_t)f * TH3 + t, 0, 0);
  __builtin_prefetch(xT_b + (size_t)f * TH3 + t, 0, 0);
  __syncthreads();

  const int wv   = t >> 5;        // wave 0..5
  const int l    = t & 31;
  const int lh   = l & 15;        // lane within half-wave
  const int rsel = l >> 4;        // which row of the pair

  const float4* __restrict__ ubase = (const float4*)(U_w + (size_t)f * TH3 * HH);
  const float4 hv = ((const float4*)sHt)[lh];   // h = 4*lh .. 4*lh+3, invariant

#pragma unroll 4
  for (int j = 0; j < 16; ++j) {
    const int p = wv * 16 + j;              // row pair (2p, 2p+1), p in 0..95
    float4 u = ubase[p * 32 + l];           // coalesced 512B per wave instr
    float s = fmaf(u.x, hv.x, fmaf(u.y, hv.y, fmaf(u.z, hv.z, u.w * hv.w)));
    s += __shfl_xor(s, 8);
    s += __shfl_xor(s, 4);
    s += __shfl_xor(s, 2);
    s += __shfl_xor(s, 1);
    if (lh == 0) sU[2 * p + rsel] = s;
  }
  sX[t] = fmaf(xT_w[(size_t)f * TH3 + t], X[f], xT_b[(size_t)f * TH3 + t]);
  __syncthreads();

  if (t < HH) {
    float z   = 1.0f / (1.0f + expf(-(sX[t]        + sU[t])));
    float r   = 1.0f / (1.0f + expf(-(sX[HH + t]   + sU[HH + t])));
    float htl = tanhf(sX[2 * HH + t] + r * sU[2 * HH + t]);
    float hg  = fmaf(z, sHt[t] - htl, htl);          // z*Ht + (1-z)*h_tilde
    bool  m   = (mask[f] != 0);
    Hout[(size_t)f * HH + t] = m ? hg : 0.0f;        // == h_gate * mf
  }
}

// ---------------------------------------------------------------------------
// K2a: deterministic column reduction of Hout (= h_gate*mf) into 256 partials,
// plus per-block mask counts (so the single head wave doesn't scan 16KB).
// Block b sums features [64b, 64b+64); coalesced reads, fixed add order.
// ---------------------------------------------------------------------------
__global__ __launch_bounds__(256) void col_reduce(
    const float* __restrict__ Hout, const unsigned char* __restrict__ mask,
    float* __restrict__ partials, int* __restrict__ cnt_part)
{
  const int b  = blockIdx.x;     // 256 blocks
  const int t  = threadIdx.x;    // 256 threads
  const int h  = t & (HH - 1);
  const int fo = t >> 6;         // 0..3
  __shared__ float sP[4][HH];
  __shared__ int   sCnt;
  if (t == 0) sCnt = 0;
  float acc = 0.0f;
#pragma unroll
  for (int i = 0; i < 16; ++i) {
    int f = b * 64 + i * 4 + fo;
    acc += Hout[(size_t)f * HH + h];
  }
  sP[fo][h] = acc;
  __syncthreads();
  if (t < 64 && mask[b * 64 + t] != 0) atomicAdd(&sCnt, 1);
  if (fo == 0)
    partials[(size_t)b * HH + h] = (sP[0][h] + sP[1][h]) + (sP[2][h] + sP[3][h]);
  __syncthreads();
  if (t == 0) cnt_part[b] = sCnt;
}

// ---------------------------------------------------------------------------
// K2b: single-wave head. Sums partials -> agg, sums counts, normalizes,
// then hid = relu(agg @ W1 + b1) via V_WMMA_F32_16X16X4_F32 (f16 fallback),
// logits = hid @ W2 + b2, softmax -> out[0..1]. Writes count for fixup.
// A is row-replicated agg (16x4 A layout: VGPR0 = K0/K2, VGPR1 = K1/K3 by
// half-wave); B tiles W1 (4x16); C accumulates over 16 K-chunks, 4 N-tiles.
// ---------------------------------------------------------------------------
__global__ __launch_bounds__(32) void head(
    const float* __restrict__ partials, const int* __restrict__ cnt_part,
    const float* __restrict__ W1, const float* __restrict__ b1,
    const float* __restrict__ W2, const float* __restrict__ b2,
    float* __restrict__ out, int* __restrict__ ws_cnt)
{
  const int lane = threadIdx.x;
  __shared__ float sAgg[HH];
  __shared__ float sHid[HH];
  __shared__ int   sC[32];
  __shared__ int   sTot;

  int c = 0;
  for (int i = lane; i < 256; i += 32) c += cnt_part[i];
  sC[lane] = c;
  for (int j = lane; j < HH; j += 32) {
    float s = 0.0f;
    for (int b = 0; b < 256; ++b) s += partials[b * HH + j];
    sAgg[j] = s;
  }
  __syncthreads();
  if (lane == 0) {
    int tot = 0;
    for (int r = 0; r < 32; ++r) tot += sC[r];
    sTot = tot;
    ws_cnt[0] = tot;
  }
  __syncthreads();
  float denom = fmaxf((float)sTot, 1.0f);
  for (int j = lane; j < HH; j += 32) sAgg[j] = sAgg[j] / denom;
  __syncthreads();

  const int half = lane >> 4;   // 0: lanes 0-15, 1: lanes 16-31
  const int l16  = lane & 15;

#if __has_builtin(__builtin_amdgcn_wmma_f32_16x16x4_f32)
  for (int nt = 0; nt < 4; ++nt) {
    v8f cacc = {};
#pragma unroll
    for (int kb = 0; kb < 16; ++kb) {
      const int k0 = kb * 4 + half * 2;
      v2f a, bb;
      a.x  = sAgg[k0];
      a.y  = sAgg[k0 + 1];
      bb.x = W1[k0 * HH + nt * 16 + l16];
      bb.y = W1[(k0 + 1) * HH + nt * 16 + l16];
      cacc = __builtin_amdgcn_wmma_f32_16x16x4_f32(
          false, a, false, bb, (short)0, cacc, false, false);
    }
    // D VGPR0: lanes 0-15 -> M=0,N=l16 (all rows equal: A rows replicated)
    if (half == 0) sHid[nt * 16 + l16] = fmaxf(cacc[0] + b1[nt * 16 + l16], 0.0f);
  }
#else
  for (int nt = 0; nt < 4; ++nt) {
    v8f cacc = {};
    for (int kc = 0; kc < 2; ++kc) {   // K = 64 = 2 x 32
      v16h a, bb;
#pragma unroll
      for (int v = 0; v < 8; ++v) {
        const int kk = kc * 32 + ((v < 4) ? (2 * v) : (16 + 2 * (v - 4))) + half * 8;
        a[2 * v]      = (_Float16)sAgg[kk];
        a[2 * v + 1]  = (_Float16)sAgg[kk + 1];
        bb[2 * v]     = (_Float16)W1[kk * HH + nt * 16 + l16];
        bb[2 * v + 1] = (_Float16)W1[(kk + 1) * HH + nt * 16 + l16];
      }
      cacc = __builtin_amdgcn_wmma_f32_16x16x32_f16(
          false, a, false, bb, (short)0, cacc, false, false);
    }
    if (half == 0) sHid[nt * 16 + l16] = fmaxf(cacc[0] + b1[nt * 16 + l16], 0.0f);
  }
#endif
  __syncthreads();

  if (lane == 0) {
    float l0 = b2[0], l1 = b2[1];
    for (int n = 0; n < HH; ++n) {
      l0 = fmaf(sHid[n], W2[n * 2 + 0], l0);
      l1 = fmaf(sHid[n], W2[n * 2 + 1], l1);
    }
    float mx = fmaxf(l0, l1);
    float e0 = expf(l0 - mx), e1 = expf(l1 - mx);
    float inv = 1.0f / (e0 + e1);
    out[0] = e0 * inv;
    out[1] = e1 * inv;
  }
}

// ---------------------------------------------------------------------------
// K3: rare fixup — if no feature is masked, H_curr = Ht.
// ---------------------------------------------------------------------------
__global__ __launch_bounds__(256) void fixup(
    const int* __restrict__ ws_cnt, const float* __restrict__ Ht,
    float* __restrict__ Hout)
{
  if (ws_cnt[0] != 0) return;
  const int stride = gridDim.x * blockDim.x;
  for (int i = blockIdx.x * blockDim.x + threadIdx.x; i < NF * HH; i += stride)
    Hout[i] = Ht[i];
}

// ---------------------------------------------------------------------------
// Inputs (setup_inputs order): 0:tim 1:X 2:X_hap(unused) 3:mask 4:Ht
// 5:xT_w 6:xT_b 7:U_w 8:W1 9:b1 10:W2 11:b2
// d_out: [pred(2), H_curr(16384*64)] fp32.
// ws: partials 256*64 f32 (64KB) @0, cnt_part 256 ints @64KB, count @64KB+1KB.
// ---------------------------------------------------------------------------
extern "C" void kernel_launch(void* const* d_in, const int* in_sizes, int n_in,
                              void* d_out, int out_size, void* d_ws, size_t ws_size,
                              hipStream_t stream) {
  (void)in_sizes; (void)n_in; (void)out_size; (void)ws_size;
  const float*         X    = (const float*)d_in[1];
  const unsigned char* mask = (const unsigned char*)d_in[3];  // jax bool = 1 byte
  const float*         Ht   = (const float*)d_in[4];
  const float*         xT_w = (const float*)d_in[5];
  const float*         xT_b = (const float*)d_in[6];
  const float*         U_w  = (const float*)d_in[7];
  const float*         W1   = (const float*)d_in[8];
  const float*         b1   = (const float*)d_in[9];
  const float*         W2   = (const float*)d_in[10];
  const float*         b2   = (const float*)d_in[11];

  float* out      = (float*)d_out;
  float* Hout     = out + 2;
  float* partials = (float*)d_ws;
  int*   cnt_part = (int*)((char*)d_ws + 256 * HH * sizeof(float));
  int*   ws_cnt   = (int*)((char*)d_ws + 256 * HH * sizeof(float) + 256 * sizeof(int));

  gru_stream<<<NF, TH3, 0, stream>>>(X, mask, Ht, xT_w, xT_b, U_w, Hout);
  col_reduce<<<256, 256, 0, stream>>>(Hout, mask, partials, cnt_part);
  head<<<1, 32, 0, stream>>>(partials, cnt_part, W1, b1, W2, b2, out, ws_cnt);
  fixup<<<256, 256, 0, stream>>>(ws_cnt, Ht, Hout);
}
